// FusedQnA1d_87282325389585
// MI455X (gfx1250) — matrix operand order repaired and checked by reading, not compiled, not run on hardware
//
#include <hip/hip_runtime.h>
#include <math.h>

#define BSZ   8
#define CINC  256
#define HIDC  256
#define NHEAD 8
#define HDIM  32
#define KW    31
#define PADW  15
#define WLEN  16384
#define WT    64      // GEMM w-tile
#define XSTR  272     // LDS transposed-tile row stride in halves (multiple of 16)
#define CT    128     // conv w-tile

typedef __attribute__((ext_vector_type(16))) _Float16 v16h;
typedef __attribute__((ext_vector_type(8)))  _Float16 v8h;
typedef __attribute__((ext_vector_type(4)))  _Float16 v4h;
typedef __attribute__((ext_vector_type(8)))  float    v8f;

// ---------------- workspace layout (bytes) ----------------
#define WS_WVH    ((size_t)0)                     // 256*256 f16  = 131072
#define WS_WOH    ((size_t)131072)                // 256*256 f16  = 131072
#define WS_QWKT   ((size_t)262144)                // 8*256 f16    = 4096
#define WS_QWKB   ((size_t)266240)                // 8 f32
#define WS_RPEE   ((size_t)266496)                // 8*31 f32
#define WS_COSTE  ((size_t)267520)                // 8*8*16384 f32 = 4194304
#define WS_VC     ((size_t)4461824)               // 8*256*16384 f16 = 67108864
#define WS_OUTP   ((size_t)71570688)              // 8*256*16384 f16 = 67108864
// total ~138.7 MB

// ============================================================
// prep: q-normalize, qWk fold, rpe exp, weight f32->f16 convert
// ============================================================
__global__ __launch_bounds__(256) void prep_kernel(
    const float* __restrict__ query, const float* __restrict__ Wk,
    const float* __restrict__ Wk_bias, const float* __restrict__ rpe,
    const float* __restrict__ to_v_w, const float* __restrict__ to_out_w,
    _Float16* __restrict__ WvH, _Float16* __restrict__ WoH,
    _Float16* __restrict__ qWkT, float* __restrict__ qWk_b,
    float* __restrict__ rpe_exp)
{
  const int t = threadIdx.x;
  if (blockIdx.x < 64) {
    // convert one of the two 256x256 weight matrices to f16
    const float* src; _Float16* dst; int bb = blockIdx.x;
    if (bb < 32) { src = to_v_w;  dst = WvH; }
    else         { src = to_out_w; dst = WoH; bb -= 32; }
    const int base = (bb * 256 + t) * 8;
#pragma unroll
    for (int j = 0; j < 8; j += 4) {
      float4 a = *(const float4*)(src + base + j);
      v4h o = { (_Float16)a.x, (_Float16)a.y, (_Float16)a.z, (_Float16)a.w };
      *(v4h*)(dst + base + j) = o;
    }
    return;
  }
  // block 64: small prep
  __shared__ float qn[HIDC];
  __shared__ float nrm[NHEAD];
  if (t < NHEAD) {
    float s = 0.f;
    for (int d = 0; d < HDIM; ++d) { float v = query[t * HDIM + d]; s += v * v; }
    nrm[t] = 1.f / ((sqrtf(s) + 1e-6f) * sqrtf((float)HDIM));
  }
  __syncthreads();
  qn[t] = query[t] * nrm[t >> 5];
  __syncthreads();
  // qWkT[h][D] = sum_d qn[h*32+d] * Wk[D][h*32+d]   (D = t)
  for (int h = 0; h < NHEAD; ++h) {
    float acc = 0.f;
    for (int d = 0; d < HDIM; ++d) acc += qn[h * HDIM + d] * Wk[t * HIDC + h * HDIM + d];
    qWkT[h * CINC + t] = (_Float16)acc;
  }
  if (t < NHEAD) {
    float acc = 0.f;
    for (int d = 0; d < HDIM; ++d) acc += Wk_bias[t * HDIM + d] * qn[t * HDIM + d];
    qWk_b[t] = acc;
  }
  if (t < NHEAD * KW) rpe_exp[t] = expf(rpe[t]);
}

// ============================================================
// qkv: cost_exp (VALU) + v GEMM (WMMA) + vc = cost_exp * (v+b)
// grid = BSZ * (WLEN/WT) blocks of 256
// ============================================================
__global__ __launch_bounds__(256) void qkv_kernel(
    const float* __restrict__ x, const _Float16* __restrict__ WvH,
    const _Float16* __restrict__ qWkT, const float* __restrict__ qWk_b,
    const float* __restrict__ to_v_b, float* __restrict__ cost_exp_g,
    _Float16* __restrict__ vc)
{
  __shared__ __attribute__((aligned(32))) _Float16 xT[WT][XSTR];     // 34 KB
  __shared__ __attribute__((aligned(32))) _Float16 qk[NHEAD][CINC];  // 4 KB
  __shared__ float costL[NHEAD][WT];                                 // 2 KB
  __shared__ float vbL[HIDC];                                        // 1 KB

  const int t  = threadIdx.x;
  const int b  = blockIdx.x >> 8;              // WLEN/WT = 256 tiles
  const int w0 = (blockIdx.x & 255) * WT;
  const float* xb = x + (size_t)b * CINC * WLEN;

  // stage qWkT (f16) and v-bias
  ((v8h*)&qk[0][0])[t] = ((const v8h*)qWkT)[t];   // 2048 halves
  vbL[t] = to_v_b[t];

  // stage x tile fp32 -> f16, transposed: xT[w][d]
#pragma unroll
  for (int i = 0; i < 16; ++i) {
    int s  = t + 256 * i;         // over [256 d][16 quads]
    int d  = s >> 4;
    int wq = (s & 15) << 2;
    float4 a = *(const float4*)(xb + (size_t)d * WLEN + w0 + wq);
    xT[wq + 0][d] = (_Float16)a.x;
    xT[wq + 1][d] = (_Float16)a.y;
    xT[wq + 2][d] = (_Float16)a.z;
    xT[wq + 3][d] = (_Float16)a.w;
  }
  __syncthreads();

  // ---- cost_exp: 8 heads x 64 w, each thread does 2 heads for one w ----
  {
    const int w  = t & 63;
    const int h0 = (t >> 6) << 1;
    float a0 = 0.f, a1 = 0.f;
    for (int d = 0; d < CINC; d += 8) {
      v8h xv = *(const v8h*)&xT[w][d];
      v8h k0 = *(const v8h*)&qk[h0][d];
      v8h k1 = *(const v8h*)&qk[h0 + 1][d];
#pragma unroll
      for (int j = 0; j < 8; ++j) {
        float xf = (float)xv[j];
        a0 += xf * (float)k0[j];
        a1 += xf * (float)k1[j];
      }
    }
    a0 = __expf(a0 + qWk_b[h0]);
    a1 = __expf(a1 + qWk_b[h0 + 1]);
    costL[h0][w]     = a0;
    costL[h0 + 1][w] = a1;
    cost_exp_g[((size_t)b * NHEAD + h0) * WLEN + w0 + w]     = a0;
    cost_exp_g[((size_t)b * NHEAD + h0 + 1) * WLEN + w0 + w] = a1;
  }
  __syncthreads();

  // ---- WMMA GEMM: v[o][w] = sum_d WvH[o][d] * xT[w][d] ----
  const int lane = t & 31;
  const int wv   = t >> 5;
  const int r0   = wv * 32;
  const int lo   = lane & 15;
  const int hi   = lane >> 4;

  v8f acc[2][4] = {};
  for (int k0 = 0; k0 < CINC; k0 += 32) {
    union { v16h v; v8h h[2]; } A[2];
#pragma unroll
    for (int rt = 0; rt < 2; ++rt) {
      const _Float16* arow = WvH + (size_t)(r0 + rt * 16 + lo) * CINC + k0 + hi * 16;
      A[rt].h[0] = *(const v8h*)(arow);
      A[rt].h[1] = *(const v8h*)(arow + 8);
    }
    v16h Bf[4];
#pragma unroll
    for (int ct = 0; ct < 4; ++ct)
      Bf[ct] = *(const v16h*)&xT[ct * 16 + lo][k0 + hi * 16];
#pragma unroll
    for (int rt = 0; rt < 2; ++rt)
#pragma unroll
      for (int ct = 0; ct < 4; ++ct)
        acc[rt][ct] = __builtin_amdgcn_wmma_f32_16x16x32_f16(
            false, A[rt].v, false, Bf[ct], (short)0, acc[rt][ct], false, false);
  }

  // ---- epilogue: vc = cost_exp * (v + bias), store f16 ----
#pragma unroll
  for (int rt = 0; rt < 2; ++rt) {
#pragma unroll
    for (int ct = 0; ct < 4; ++ct) {
      const int w = ct * 16 + lo;
#pragma unroll
      for (int r = 0; r < 8; ++r) {
        const int o = r0 + rt * 16 + hi * 8 + r;
        float v = acc[rt][ct][r] + vbL[o];
        float ce = costL[o >> 5][w];
        vc[((size_t)b * HIDC + o) * WLEN + w0 + w] = (_Float16)(v * ce);
      }
    }
  }
}

// ============================================================
// conv: depthwise K=31 correlation over w on vc and cost_exp,
// divide -> out_pre (f16). grid = BSZ*NHEAD*(WLEN/CT)
// ============================================================
__global__ __launch_bounds__(256) void conv_kernel(
    const float* __restrict__ cost_exp_g, const _Float16* __restrict__ vc,
    const float* __restrict__ rpe_exp, _Float16* __restrict__ out_pre)
{
  __shared__ float ce[CT + 30];
  __shared__ float rinv[CT];
  __shared__ float rk[KW];
  __shared__ _Float16 vcl[HDIM][CT + 32];   // padded stride

  const int t  = threadIdx.x;
  const int nt = WLEN / CT;                  // 128
  const int wt = blockIdx.x % nt;
  const int h  = (blockIdx.x / nt) % NHEAD;
  const int b  = blockIdx.x / (nt * NHEAD);
  const int w0 = wt * CT;

  if (t < KW) rk[t] = rpe_exp[h * KW + t];
  const float* ceg = cost_exp_g + ((size_t)b * NHEAD + h) * WLEN;
  for (int i = t; i < CT + 30; i += 256) {
    int w = w0 + i - PADW;
    ce[i] = (w >= 0 && w < WLEN) ? ceg[w] : 0.f;
  }
  const _Float16* vcg = vc + ((size_t)b * HIDC + h * HDIM) * WLEN;
  for (int i = t; i < HDIM * (CT + 30); i += 256) {
    int c = i / (CT + 30);
    int j = i % (CT + 30);
    int w = w0 + j - PADW;
    vcl[c][j] = (w >= 0 && w < WLEN) ? vcg[(size_t)c * WLEN + w] : (_Float16)0.f;
  }
  __syncthreads();

  if (t < CT) {
    float s = 0.f;
#pragma unroll
    for (int k = 0; k < KW; ++k) s += ce[t + k] * rk[k];
    rinv[t] = 1.f / s;
  }
  __syncthreads();

#pragma unroll
  for (int i = 0; i < 16; ++i) {
    int s = t + 256 * i;          // over 32c x 128w
    int c = s >> 7;
    int w = s & 127;
    float a = 0.f;
#pragma unroll
    for (int k = 0; k < KW; ++k) a += (float)vcl[c][w + k] * rk[k];
    out_pre[((size_t)(b * HIDC + h * HDIM + c)) * WLEN + w0 + w] = (_Float16)(a * rinv[w]);
  }
}

// ============================================================
// out: to_out GEMM (WMMA) + bias, fp32 output
// grid = BSZ * (WLEN/WT)
// ============================================================
__global__ __launch_bounds__(256) void out_kernel(
    const _Float16* __restrict__ out_pre, const _Float16* __restrict__ WoH,
    const float* __restrict__ to_out_b, float* __restrict__ out)
{
  __shared__ __attribute__((aligned(32))) _Float16 xT[WT][XSTR];
  __shared__ float obL[HIDC];

  const int t  = threadIdx.x;
  const int b  = blockIdx.x >> 8;
  const int w0 = (blockIdx.x & 255) * WT;
  obL[t] = to_out_b[t];

  const _Float16* pb = out_pre + (size_t)b * HIDC * WLEN;
#pragma unroll
  for (int i = 0; i < 8; ++i) {
    int s = t + 256 * i;           // over [256 d][8 octets]
    int d = s >> 3;
    int c = (s & 7) << 3;
    v8h a = *(const v8h*)(pb + (size_t)d * WLEN + w0 + c);
#pragma unroll
    for (int j = 0; j < 8; ++j) xT[c + j][d] = a[j];
  }
  __syncthreads();

  const int lane = t & 31;
  const int wv   = t >> 5;
  const int r0   = wv * 32;
  const int lo   = lane & 15;
  const int hi   = lane >> 4;

  v8f acc[2][4] = {};
  for (int k0 = 0; k0 < HIDC; k0 += 32) {
    union { v16h v; v8h h[2]; } A[2];
#pragma unroll
    for (int rt = 0; rt < 2; ++rt) {
      const _Float16* arow = WoH + (size_t)(r0 + rt * 16 + lo) * HIDC + k0 + hi * 16;
      A[rt].h[0] = *(const v8h*)(arow);
      A[rt].h[1] = *(const v8h*)(arow + 8);
    }
    v16h Bf[4];
#pragma unroll
    for (int ct = 0; ct < 4; ++ct)
      Bf[ct] = *(const v16h*)&xT[ct * 16 + lo][k0 + hi * 16];
#pragma unroll
    for (int rt = 0; rt < 2; ++rt)
#pragma unroll
      for (int ct = 0; ct < 4; ++ct)
        acc[rt][ct] = __builtin_amdgcn_wmma_f32_16x16x32_f16(
            false, A[rt].v, false, Bf[ct], (short)0, acc[rt][ct], false, false);
  }

#pragma unroll
  for (int rt = 0; rt < 2; ++rt) {
#pragma unroll
    for (int ct = 0; ct < 4; ++ct) {
      const int w = ct * 16 + lo;
#pragma unroll
      for (int r = 0; r < 8; ++r) {
        const int o = r0 + rt * 16 + hi * 8 + r;
        out[((size_t)b * HIDC + o) * WLEN + w0 + w] = acc[rt][ct][r] + obL[o];
      }
    }
  }
}

// ============================================================
extern "C" void kernel_launch(void* const* d_in, const int* in_sizes, int n_in,
                              void* d_out, int out_size, void* d_ws, size_t ws_size,
                              hipStream_t stream) {
  const float* x        = (const float*)d_in[0];
  const float* query    = (const float*)d_in[1];
  const float* Wk       = (const float*)d_in[2];
  const float* Wk_bias  = (const float*)d_in[3];
  const float* rpe      = (const float*)d_in[4];
  const float* to_v_w   = (const float*)d_in[5];
  const float* to_v_b   = (const float*)d_in[6];
  const float* to_out_w = (const float*)d_in[7];
  const float* to_out_b = (const float*)d_in[8];
  float* out = (float*)d_out;

  char* ws = (char*)d_ws;
  _Float16* WvH      = (_Float16*)(ws + WS_WVH);
  _Float16* WoH      = (_Float16*)(ws + WS_WOH);
  _Float16* qWkT     = (_Float16*)(ws + WS_QWKT);
  float*    qWk_b    = (float*)   (ws + WS_QWKB);
  float*    rpe_exp  = (float*)   (ws + WS_RPEE);
  float*    cost_exp = (float*)   (ws + WS_COSTE);
  _Float16* vc       = (_Float16*)(ws + WS_VC);
  _Float16* out_pre  = (_Float16*)(ws + WS_OUTP);

  prep_kernel<<<65, 256, 0, stream>>>(query, Wk, Wk_bias, rpe, to_v_w, to_out_w,
                                      WvH, WoH, qWkT, qWk_b, rpe_exp);
  qkv_kernel<<<BSZ * (WLEN / WT), 256, 0, stream>>>(x, WvH, qWkT, qWk_b, to_v_b,
                                                    cost_exp, vc);
  conv_kernel<<<BSZ * NHEAD * (WLEN / CT), 256, 0, stream>>>(cost_exp, vc, rpe_exp,
                                                             out_pre);
  out_kernel<<<BSZ * (WLEN / WT), 256, 0, stream>>>(out_pre, WoH, to_out_b, out);
}
